// mod_slg2_v2_5282809774454
// MI455X (gfx1250) — matrix operand order repaired and checked by reading, not compiled, not run on hardware
//
#include <hip/hip_runtime.h>
#include <hip/hip_bf16.h>
#include <math.h>

// ---------------------------------------------------------------------------
// MI455X (gfx1250, wave32) implementation.
// GEMMs run on the matrix pipe via v_wmma_f32_16x16x32_f16 (f32 accumulate).
// Weights are pre-packed once into fragment-major f16 (one 32B load / B-frag).
// FFN chains are fused per 16-row tile; the GELU intermediate is transposed
// C-layout -> A-layout through per-wave LDS as f16 (s_wait_dscnt fence).
// ---------------------------------------------------------------------------

typedef _Float16 f16;
typedef __attribute__((ext_vector_type(16))) _Float16 v16h;
typedef __attribute__((ext_vector_type(8)))  float    v8f;

#define BN_INV 0.9999950000374997f   // 1/sqrt(1+1e-5)

__device__ __forceinline__ float gelu_exact(float x) {
  return 0.5f * x * (1.0f + erff(x * 0.70710678118654752f));
}

// A-fragment (16x32 f16) from an f32 row: lane holds row (lane&15); lanes 0-15
// take K {k0..k0+7, k0+16..k0+23}, lanes 16-31 the +8 shifted halves.
__device__ __forceinline__ v16h a_frag_f32(const float* row, int k0, int hi) {
  v16h a;
  const float* p = row + k0 + 8 * hi;
#pragma unroll
  for (int j = 0; j < 8; ++j) a[j] = (f16)p[j];
#pragma unroll
  for (int j = 0; j < 8; ++j) a[8 + j] = (f16)p[16 + j];
  return a;
}

// A-fragment from an f16 row (LDS); 16B-aligned contiguous halves.
__device__ __forceinline__ v16h a_frag_h16(const f16* rowp, int k0, int hi) {
  const f16* p = rowp + k0 + 8 * hi;
  v16h a;
#pragma unroll
  for (int j = 0; j < 8; ++j) a[j] = p[j];
#pragma unroll
  for (int j = 0; j < 8; ++j) a[8 + j] = p[16 + j];
  return a;
}

// Pre-packed B fragment: one 32-byte per-lane load.
__device__ __forceinline__ v16h b_frag_pk(const f16* Wp, int frag, int lane) {
  return ((const v16h*)Wp)[frag * 32 + lane];
}

__device__ __forceinline__ v8f wmma16(v16h a, v16h b, v8f c) {
  return __builtin_amdgcn_wmma_f32_16x16x32_f16(false, a, false, b, (short)0, c,
                                                false, false);
}

// ---------------------------------------------------------------------------
// Pack W[K][Nn] (f32, row-major) into fragment-major f16.
// frag index = ks*NT + nt; per-lane 16 halves follow the 32x16 B layout:
// VGPR p holds rows (ks*32 + 8*hi + 2p + (p>=4 ? 8 : 0), +1), col = nt*16 + (lane&15).
// Rows >= K are zero-filled (handles K=16 edge-linear padding).
// ---------------------------------------------------------------------------
__global__ void k_pack_b(const float* __restrict__ W, f16* __restrict__ Wp,
                         int K, int Nn) {
  int NT = Nn >> 4, KS = (K + 31) >> 5;
  int total = KS * NT * 32;
  int idx = blockIdx.x * blockDim.x + threadIdx.x;
  if (idx >= total) return;
  int lane = idx & 31, fr = idx >> 5;
  int nt = fr % NT, ks = fr / NT;
  int m = lane & 15, hi = lane >> 4;
  int n = nt * 16 + m;
  v16h t;
#pragma unroll
  for (int r = 0; r < 8; ++r) {
    int k0 = ks * 32 + 8 * hi + 2 * r + (r >= 4 ? 8 : 0);
    t[2 * r]     = (k0 < K)     ? (f16)W[(size_t)k0 * Nn + n]       : (f16)0.0f;
    t[2 * r + 1] = (k0 + 1 < K) ? (f16)W[(size_t)(k0 + 1) * Nn + n] : (f16)0.0f;
  }
  ((v16h*)Wp)[idx] = t;
}

// ---------------------------------------------------------------------------
// Plain GEMM: O[M x 128] = A[M x 128] @ W[128 x 128]   (M % 16 == 0)
// ---------------------------------------------------------------------------
__global__ void k_gemm128(const float* __restrict__ A, const f16* __restrict__ Wp,
                          float* __restrict__ O, int M) {
  int wave = threadIdx.x >> 5, lane = threadIdx.x & 31;
  int tile = blockIdx.x * 8 + wave;
  if (tile * 16 >= M) return;
  int m = lane & 15, hi = lane >> 4;
  const float* arow = A + (size_t)(tile * 16 + m) * 128;
  v16h af[4];
#pragma unroll
  for (int ks = 0; ks < 4; ++ks) af[ks] = a_frag_f32(arow, ks * 32, hi);
#pragma unroll
  for (int nt = 0; nt < 8; ++nt) {
    v8f acc = {};
#pragma unroll
    for (int ks = 0; ks < 4; ++ks)
      acc = wmma16(af[ks], b_frag_pk(Wp, ks * 8 + nt, lane), acc);
    float* op = O + (size_t)(tile * 16 + hi * 8) * 128 + nt * 16 + m;
#pragma unroll
    for (int p = 0; p < 8; ++p) op[(size_t)p * 128] = acc[p];
  }
}

// ---------------------------------------------------------------------------
// H0_edge = edge_attr[M x 16] @ W[16 x 128] + b  (zero-pad baked into Wp)
// ---------------------------------------------------------------------------
__global__ void k_edge_linear(const float* __restrict__ EA,
                              const f16* __restrict__ Wp,
                              const float* __restrict__ bias,
                              float* __restrict__ O, int M) {
  int wave = threadIdx.x >> 5, lane = threadIdx.x & 31;
  int tile = blockIdx.x * 8 + wave;
  if (tile * 16 >= M) return;
  int m = lane & 15, hi = lane >> 4;
  const float* arow = EA + (size_t)(tile * 16 + m) * 16;
  v16h a;
#pragma unroll
  for (int j = 0; j < 8; ++j) a[j] = (f16)arow[8 * hi + j];  // K = 8*hi + j < 16
#pragma unroll
  for (int j = 0; j < 8; ++j) a[8 + j] = (f16)0.0f;          // K >= 16 -> 0
#pragma unroll
  for (int nt = 0; nt < 8; ++nt) {
    int n = nt * 16 + m;
    v8f acc = {};
    acc = wmma16(a, b_frag_pk(Wp, nt, lane), acc);
    float bv = bias[n];
    float* op = O + (size_t)(tile * 16 + hi * 8) * 128 + n;
#pragma unroll
    for (int p = 0; p < 8; ++p) op[(size_t)p * 128] = acc[p] + bv;
  }
}

// ---------------------------------------------------------------------------
// ff_ne fused: h_edge[e] = 0.5*(gelu(bn(c1@W0+b0)) + gelu(bn(c2@W0+b0)))@W1 + b1
// c1=[H0_u,H0_e,H0_v], c2=[H0_v,H0_e,H0_u]; W0 split in 3 128x64 blocks.
// W0p frags: [12 ks][4 nt]; W1p frags: [2 ks][4 nt].
// ---------------------------------------------------------------------------
#define ERS 72   // LDS row stride (halves) for 64-wide intermediate
__global__ void k_edge_ffn(const float* __restrict__ H0, const float* __restrict__ H0e,
                           const int* __restrict__ ueu, const int* __restrict__ uev,
                           const f16* __restrict__ W0p, const float* __restrict__ b0,
                           const float* __restrict__ g0, const float* __restrict__ be0,
                           const f16* __restrict__ W1p, const float* __restrict__ b1,
                           float* __restrict__ Oh, int M) {
  extern __shared__ f16 smem[];
  int wave = threadIdx.x >> 5, lane = threadIdx.x & 31;
  int tile = blockIdx.x * 8 + wave;
  if (tile * 16 >= M) return;
  int m = lane & 15, hi = lane >> 4;
  f16* gbuf = smem + wave * 16 * ERS;  // per-wave [16][ERS] f16
  int row = tile * 16 + m;
  int u = ueu[row], v = uev[row];
  const float* urow = H0 + (size_t)u * 128;
  const float* vrow = H0 + (size_t)v * 128;
  const float* erow = H0e + (size_t)row * 128;
  v16h uf[4], vf[4], ef[4];
#pragma unroll
  for (int ks = 0; ks < 4; ++ks) {
    uf[ks] = a_frag_f32(urow, ks * 32, hi);
    vf[ks] = a_frag_f32(vrow, ks * 32, hi);
    ef[ks] = a_frag_f32(erow, ks * 32, hi);
  }
#pragma unroll
  for (int nt = 0; nt < 4; ++nt) {
    int n = nt * 16 + m;
    v8f a1 = {}, a2 = {};
#pragma unroll
    for (int ks = 0; ks < 4; ++ks) {
      v16h bA = b_frag_pk(W0p, ks * 4 + nt, lane);        // W0 rows   0..127
      v16h bB = b_frag_pk(W0p, (ks + 4) * 4 + nt, lane);  // W0 rows 128..255
      v16h bC = b_frag_pk(W0p, (ks + 8) * 4 + nt, lane);  // W0 rows 256..383
      a1 = wmma16(uf[ks], bA, a1);  a2 = wmma16(vf[ks], bA, a2);
      a1 = wmma16(ef[ks], bB, a1);  a2 = wmma16(ef[ks], bB, a2);
      a1 = wmma16(vf[ks], bC, a1);  a2 = wmma16(uf[ks], bC, a2);
    }
    float bb = b0[n], gg = g0[n] * BN_INV, be = be0[n];
#pragma unroll
    for (int p = 0; p < 8; ++p) {
      float x1 = (a1[p] + bb) * gg + be;
      float x2 = (a2[p] + bb) * gg + be;
      gbuf[(p + 8 * hi) * ERS + n] = (f16)(gelu_exact(x1) + gelu_exact(x2));
    }
  }
  asm volatile("s_wait_dscnt 0x0" ::: "memory");  // LDS RAW fence (per-wave)
  v16h gf[2];
#pragma unroll
  for (int ks = 0; ks < 2; ++ks) gf[ks] = a_frag_h16(gbuf + m * ERS, ks * 32, hi);
#pragma unroll
  for (int nt = 0; nt < 4; ++nt) {
    int n = nt * 16 + m;
    v8f acc = {};
    acc = wmma16(gf[0], b_frag_pk(W1p, nt, lane), acc);
    acc = wmma16(gf[1], b_frag_pk(W1p, 4 + nt, lane), acc);
    float bv = b1[n];
    float* op = Oh + (size_t)(tile * 16 + hi * 8) * 64 + n;
#pragma unroll
    for (int p = 0; p < 8; ++p) op[(size_t)p * 64] = 0.5f * acc[p] + bv;
  }
}

// ---------------------------------------------------------------------------
// ff_ef fused: H1raw = 0.5*(gelu(bn([hf,he]@W0+b0)) + gelu(bn([he,hf]@W0+b0)))@W1 + b1
// W0p frags: [4 ks][8 nt] (ks 0-1 = top 64 rows, ks 2-3 = bottom); W1p: [4][8].
// ---------------------------------------------------------------------------
#define LRS 136  // LDS row stride (halves) for 128-wide intermediate
__global__ void k_l2_ffn(const float* __restrict__ He, const int* __restrict__ eidx,
                         const int* __restrict__ fidx,
                         const f16* __restrict__ W0p, const float* __restrict__ b0,
                         const float* __restrict__ g0, const float* __restrict__ be0,
                         const f16* __restrict__ W1p, const float* __restrict__ b1,
                         float* __restrict__ O, int M) {
  extern __shared__ f16 smem[];
  int wave = threadIdx.x >> 5, lane = threadIdx.x & 31;
  int tile = blockIdx.x * 8 + wave;
  if (tile * 16 >= M) return;
  int m = lane & 15, hi = lane >> 4;
  f16* gbuf = smem + wave * 16 * LRS;
  int row = tile * 16 + m;
  int e = eidx[row], f = fidx[row];
  const float* frow = He + (size_t)f * 64;
  const float* erow = He + (size_t)e * 64;
  v16h ffr[2], efr[2];
#pragma unroll
  for (int ks = 0; ks < 2; ++ks) {
    ffr[ks] = a_frag_f32(frow, ks * 32, hi);
    efr[ks] = a_frag_f32(erow, ks * 32, hi);
  }
#pragma unroll
  for (int nt = 0; nt < 8; ++nt) {
    int n = nt * 16 + m;
    v8f a1 = {}, a2 = {};
#pragma unroll
    for (int ks = 0; ks < 2; ++ks) {
      v16h bT = b_frag_pk(W0p, ks * 8 + nt, lane);        // rows  0..63
      v16h bB = b_frag_pk(W0p, (ks + 2) * 8 + nt, lane);  // rows 64..127
      a1 = wmma16(ffr[ks], bT, a1);  a2 = wmma16(efr[ks], bT, a2);
      a1 = wmma16(efr[ks], bB, a1);  a2 = wmma16(ffr[ks], bB, a2);
    }
    float bb = b0[n], gg = g0[n] * BN_INV, be = be0[n];
#pragma unroll
    for (int p = 0; p < 8; ++p) {
      float x1 = (a1[p] + bb) * gg + be;
      float x2 = (a2[p] + bb) * gg + be;
      gbuf[(p + 8 * hi) * LRS + n] = (f16)(gelu_exact(x1) + gelu_exact(x2));
    }
  }
  asm volatile("s_wait_dscnt 0x0" ::: "memory");
  v16h gf[4];
#pragma unroll
  for (int ks = 0; ks < 4; ++ks) gf[ks] = a_frag_h16(gbuf + m * LRS, ks * 32, hi);
#pragma unroll
  for (int nt = 0; nt < 8; ++nt) {
    int n = nt * 16 + m;
    v8f acc = {};
#pragma unroll
    for (int ks = 0; ks < 4; ++ks)
      acc = wmma16(gf[ks], b_frag_pk(W1p, ks * 8 + nt, lane), acc);
    float bv = b1[n];
    float* op = O + (size_t)(tile * 16 + hi * 8) * 128 + n;
#pragma unroll
    for (int p = 0; p < 8; ++p) op[(size_t)p * 128] = 0.5f * acc[p] + bv;
  }
}

// ---------------------------------------------------------------------------
// Elementwise / scatter helpers
// ---------------------------------------------------------------------------
__global__ void k_fill(float* p, float v, int n) {
  for (int i = blockIdx.x * blockDim.x + threadIdx.x; i < n;
       i += gridDim.x * blockDim.x) p[i] = v;
}

__global__ void k_count_deg(float* deg, const int* dst, int E) {
  for (int e = blockIdx.x * blockDim.x + threadIdx.x; e < E;
       e += gridDim.x * blockDim.x) atomicAdd(&deg[dst[e]], 1.0f);
}

__global__ void k_rsqrt_ip(float* p, int n) {
  for (int i = blockIdx.x * blockDim.x + threadIdx.x; i < n;
       i += gridDim.x * blockDim.x) p[i] = rsqrtf(p[i]);
}

// O[i][c] = bias[c] + m[i][c]*dinv[i]^2  (self-loop term)
__global__ void k_gcn_init(float* __restrict__ O, const float* __restrict__ m,
                           const float* __restrict__ dinv,
                           const float* __restrict__ bias, int N) {
  int total = N * 128;
  for (int idx = blockIdx.x * blockDim.x + threadIdx.x; idx < total;
       idx += gridDim.x * blockDim.x) {
    int i = idx >> 7, c = idx & 127;
    float di = dinv[i];
    O[idx] = bias[c] + m[idx] * di * di;
  }
}

__global__ void k_gcn_scatter(float* __restrict__ O, const float* __restrict__ m,
                              const float* __restrict__ dinv,
                              const int* __restrict__ src,
                              const int* __restrict__ dst, int E) {
  int total = E * 128;
  for (int idx = blockIdx.x * blockDim.x + threadIdx.x; idx < total;
       idx += gridDim.x * blockDim.x) {
    int e = idx >> 7, c = idx & 127;
    int s = src[e], d = dst[e];
    atomicAdd(&O[(size_t)d * 128 + c], m[(size_t)s * 128 + c] * dinv[s] * dinv[d]);
  }
}

// H1_new = 0.25*(H0[u]+H0[v]+H0[s]+H0[t]) + relu(bn1(H1raw))
__global__ void k_h1new(float* __restrict__ H1n, const float* __restrict__ H1raw,
                        const float* __restrict__ H0, const int* __restrict__ eidx,
                        const int* __restrict__ fidx, const int* __restrict__ ueu,
                        const int* __restrict__ uev, const float* __restrict__ bg,
                        const float* __restrict__ bb, int M) {
  int total = M * 128;
  for (int idx = blockIdx.x * blockDim.x + threadIdx.x; idx < total;
       idx += gridDim.x * blockDim.x) {
    int i = idx >> 7, c = idx & 127;
    int e = eidx[i], f = fidx[i];
    int u = ueu[e], v = uev[e], s = ueu[f], t = uev[f];
    float hm = 0.25f * (H0[(size_t)u * 128 + c] + H0[(size_t)v * 128 + c] +
                        H0[(size_t)s * 128 + c] + H0[(size_t)t * 128 + c]);
    float x = H1raw[idx] * (bg[c] * BN_INV) + bb[c];
    H1n[idx] = hm + fmaxf(x, 0.0f);
  }
}

// H2 <- H1_new + relu(bn2(H2))  (in place)
__global__ void k_h2new(float* __restrict__ H2, const float* __restrict__ H1n,
                        const float* __restrict__ bg, const float* __restrict__ bb,
                        int M) {
  int total = M * 128;
  for (int idx = blockIdx.x * blockDim.x + threadIdx.x; idx < total;
       idx += gridDim.x * blockDim.x) {
    int c = idx & 127;
    float x = H2[idx] * (bg[c] * BN_INV) + bb[c];
    H2[idx] = H1n[idx] + fmaxf(x, 0.0f);
  }
}

__global__ void k_pool_cnt(float* cnt, const int* eidx, const int* ueu,
                           const int* batch, int M) {
  int i = blockIdx.x * blockDim.x + threadIdx.x;
  if (i < M) atomicAdd(&cnt[batch[ueu[eidx[i]]]], 1.0f);
}

__global__ void k_pool_sum(float* __restrict__ p1, float* __restrict__ p2,
                           const float* __restrict__ H1, const float* __restrict__ H2,
                           const int* __restrict__ eidx, const int* __restrict__ ueu,
                           const int* __restrict__ batch, int M) {
  int total = M * 128;
  for (int idx = blockIdx.x * blockDim.x + threadIdx.x; idx < total;
       idx += gridDim.x * blockDim.x) {
    int i = idx >> 7, c = idx & 127;
    int b = batch[ueu[eidx[i]]];
    atomicAdd(&p1[b * 128 + c], H1[idx]);
    atomicAdd(&p2[b * 128 + c], H2[idx]);
  }
}

__global__ void k_pool_div(float* Hp, const float* p1, const float* p2,
                           const float* cnt) {
  int g = blockIdx.x, t = threadIdx.x;
  float d = fmaxf(cnt[g], 1.0f);
  Hp[g * 256 + t]       = p1[g * 128 + t] / d;
  Hp[g * 256 + 128 + t] = p2[g * 128 + t] / d;
}

// ---------------------------------------------------------------------------
// Head MLP: 64 graphs, one 128-thread block each.
// ---------------------------------------------------------------------------
__device__ float blk_reduce(float v, float* buf) {
  int t = threadIdx.x;
  buf[t] = v;  __syncthreads();
  for (int s = 64; s > 0; s >>= 1) {
    if (t < s) buf[t] += buf[t + s];
    __syncthreads();
  }
  float r = buf[0];  __syncthreads();
  return r;
}

__global__ void k_head(const float* __restrict__ Hp, const float* W0, const float* b0,
                       const float* lg0, const float* lb0, const float* W1,
                       const float* b1, const float* lg1, const float* lb1,
                       const float* W2, const float* b2, float* out) {
  __shared__ float sh[128];
  __shared__ float rbuf[128];
  int g = blockIdx.x, t = threadIdx.x;
  const float* xp = Hp + g * 256;
  float acc = b0[t];
  for (int k = 0; k < 256; ++k) acc += xp[k] * W0[k * 128 + t];
  float s1 = blk_reduce(acc, rbuf);
  float s2 = blk_reduce(acc * acc, rbuf);
  float mu = s1 / 128.0f, var = s2 / 128.0f - mu * mu;
  float y = gelu_exact((acc - mu) * rsqrtf(var + 1e-5f) * lg0[t] + lb0[t]);
  sh[t] = y;  __syncthreads();
  acc = b1[t];
  for (int k = 0; k < 128; ++k) acc += sh[k] * W1[k * 128 + t];
  s1 = blk_reduce(acc, rbuf);
  s2 = blk_reduce(acc * acc, rbuf);
  mu = s1 / 128.0f;  var = s2 / 128.0f - mu * mu;
  y = gelu_exact((acc - mu) * rsqrtf(var + 1e-5f) * lg1[t] + lb1[t]);
  __syncthreads();
  sh[t] = y;  __syncthreads();
  if (t < 8) {
    float o = b2[t];
    for (int k = 0; k < 128; ++k) o += sh[k] * W2[k * 8 + t];
    out[g * 8 + t] = o;
  }
}

// ---------------------------------------------------------------------------
// Launch
// ---------------------------------------------------------------------------
extern "C" void kernel_launch(void* const* d_in, const int* in_sizes, int n_in,
                              void* d_out, int out_size, void* d_ws, size_t ws_size,
                              hipStream_t stream) {
  (void)in_sizes; (void)n_in; (void)out_size; (void)ws_size;
  const int N = 20000, E_DIR = 320000, E_UND = 160000, N_L2 = 200000,
            E_L2 = 400000, G = 64;

  // inputs (setup_inputs dict order, params flattened in insertion order)
  const float* x        = (const float*)d_in[0];
  const float* eattr    = (const float*)d_in[1];
  const float* linW     = (const float*)d_in[2];
  const float* linB     = (const float*)d_in[3];
  const float* g1W      = (const float*)d_in[4];
  const float* g1B      = (const float*)d_in[5];
  const float* neW0     = (const float*)d_in[6];
  const float* neB0     = (const float*)d_in[7];
  const float* neG0     = (const float*)d_in[8];
  const float* neBe0    = (const float*)d_in[9];
  const float* neW1     = (const float*)d_in[10];
  const float* neB1     = (const float*)d_in[11];
  const float* efW0     = (const float*)d_in[12];
  const float* efB0     = (const float*)d_in[13];
  const float* efG0     = (const float*)d_in[14];
  const float* efBe0    = (const float*)d_in[15];
  const float* efW1     = (const float*)d_in[16];
  const float* efB1     = (const float*)d_in[17];
  const float* bn1g     = (const float*)d_in[18];
  const float* bn1b     = (const float*)d_in[19];
  const float* g2W      = (const float*)d_in[20];
  const float* g2B      = (const float*)d_in[21];
  const float* bn2g     = (const float*)d_in[22];
  const float* bn2b     = (const float*)d_in[23];
  const float* hW0      = (const float*)d_in[24];
  const float* hB0      = (const float*)d_in[25];
  const float* hLg0     = (const float*)d_in[26];
  const float* hLb0     = (const float*)d_in[27];
  const float* hW1      = (const float*)d_in[28];
  const float* hB1      = (const float*)d_in[29];
  const float* hLg1     = (const float*)d_in[30];
  const float* hLb1     = (const float*)d_in[31];
  const float* hW2      = (const float*)d_in[32];
  const float* hB2      = (const float*)d_in[33];
  const int* ei         = (const int*)d_in[34];   // [2, E_DIR]
  const int* ue         = (const int*)d_in[35];   // [2, E_UND]
  const int* l2map      = (const int*)d_in[36];   // [2, N_L2]
  const int* l2ei       = (const int*)d_in[37];   // [2, E_L2]
  const int* batch      = (const int*)d_in[38];
  const int* src = ei,        *dst = ei + E_DIR;
  const int* ueu = ue,        *uev = ue + E_UND;
  const int* eidx = l2map,    *fidx = l2map + N_L2;
  const int* l2s = l2ei,      *l2d = l2ei + E_L2;
  float* out = (float*)d_out;

  // workspace layout (floats)
  float* ws    = (float*)d_ws;
  float* H1n   = ws;                    // 25.6M  (H1_new, persistent)
  float* B2    = ws + 25600000;         // 25.6M  (H0_edge -> H1raw -> m2)
  float* B3    = ws + 51200000;         // 25.6M  (h_edge -> H2/H2_new)
  float* m1    = ws + 76800000;         // 2.56M
  float* H0    = ws + 79360000;         // 2.56M
  float* deg1  = ws + 81920000;         // 20000 (dinv in place)
  float* deg2  = ws + 81940000;         // 200000
  float* p1    = ws + 82140000;         // 8192
  float* p2    = p1 + G * 128;          // 8192
  float* cnt   = p2 + G * 128;          // 64
  float* Hp    = cnt + G;               // 16384
  // packed f16 weights (fragment-major), 32B-aligned region
  f16* wpk     = (f16*)(ws + 82200000);
  f16* g1Wp    = wpk;                   // 4ks*8nt*512 = 16384 halves
  f16* linWp   = wpk + 16384;           // 1*8*512    =  4096
  f16* neW0p   = wpk + 20480;           // 12*4*512   = 24576
  f16* neW1p   = wpk + 45056;           // 2*4*512    =  4096
  f16* efW0p   = wpk + 49152;           // 4*8*512    = 16384
  f16* efW1p   = wpk + 65536;           // 4*8*512    = 16384
  f16* g2Wp    = wpk + 81920;           // 4*8*512    = 16384

  const int T = 256;
  // ---- pack all weight matrices into fragment-major f16 ----
  k_pack_b<<<4,  T, 0, stream>>>(g1W,  g1Wp,  128, 128);
  k_pack_b<<<1,  T, 0, stream>>>(linW, linWp, 16,  128);
  k_pack_b<<<6,  T, 0, stream>>>(neW0, neW0p, 384, 64);
  k_pack_b<<<1,  T, 0, stream>>>(neW1, neW1p, 64,  64);
  k_pack_b<<<4,  T, 0, stream>>>(efW0, efW0p, 128, 128);
  k_pack_b<<<4,  T, 0, stream>>>(efW1, efW1p, 128, 128);
  k_pack_b<<<4,  T, 0, stream>>>(g2W,  g2Wp,  128, 128);
  // ---- GCN layer 1 ----
  k_fill<<<(N + T - 1) / T, T, 0, stream>>>(deg1, 1.0f, N);
  k_count_deg<<<(E_DIR + T - 1) / T, T, 0, stream>>>(deg1, dst, E_DIR);
  k_rsqrt_ip<<<(N + T - 1) / T, T, 0, stream>>>(deg1, N);
  k_gemm128<<<(N / 16 + 7) / 8, T, 0, stream>>>(x, g1Wp, m1, N);
  k_gcn_init<<<2048, T, 0, stream>>>(H0, m1, deg1, g1B, N);
  k_gcn_scatter<<<8192, T, 0, stream>>>(H0, m1, deg1, src, dst, E_DIR);
  // ---- edge linear + ff_ne ----
  k_edge_linear<<<(E_UND / 16 + 7) / 8, T, 0, stream>>>(eattr, linWp, linB, B2, E_UND);
  k_edge_ffn<<<(E_UND / 16 + 7) / 8, T, 8 * 16 * ERS * 2, stream>>>(
      H0, B2, ueu, uev, neW0p, neB0, neG0, neBe0, neW1p, neB1, B3, E_UND);
  // ---- ff_ef + H1_new ----
  k_l2_ffn<<<(N_L2 / 16 + 7) / 8, T, 8 * 16 * LRS * 2, stream>>>(
      B3, eidx, fidx, efW0p, efB0, efG0, efBe0, efW1p, efB1, B2, N_L2);
  k_h1new<<<4096, T, 0, stream>>>(H1n, B2, H0, eidx, fidx, ueu, uev, bn1g, bn1b, N_L2);
  // ---- GCN layer 2 ----
  k_fill<<<(N_L2 + T - 1) / T, T, 0, stream>>>(deg2, 1.0f, N_L2);
  k_count_deg<<<(E_L2 + T - 1) / T, T, 0, stream>>>(deg2, l2d, E_L2);
  k_rsqrt_ip<<<(N_L2 + T - 1) / T, T, 0, stream>>>(deg2, N_L2);
  k_gemm128<<<(N_L2 / 16 + 7) / 8, T, 0, stream>>>(H1n, g2Wp, B2, N_L2);
  k_gcn_init<<<4096, T, 0, stream>>>(B3, B2, deg2, g2B, N_L2);
  k_gcn_scatter<<<8192, T, 0, stream>>>(B3, B2, deg2, l2s, l2d, E_L2);
  k_h2new<<<4096, T, 0, stream>>>(B3, H1n, bn2g, bn2b, N_L2);
  // ---- pooling ----
  k_fill<<<(G * 257 + T - 1) / T, T, 0, stream>>>(p1, 0.0f, G * 128 * 2 + G);
  k_pool_cnt<<<(N_L2 + T - 1) / T, T, 0, stream>>>(cnt, eidx, ueu, batch, N_L2);
  k_pool_sum<<<4096, T, 0, stream>>>(p1, p2, H1n, B3, eidx, ueu, batch, N_L2);
  k_pool_div<<<G, 128, 0, stream>>>(Hp, p1, p2, cnt);
  // ---- head ----
  k_head<<<G, 128, 0, stream>>>(Hp, hW0, hB0, hLg0, hLb0, hW1, hB1, hLg1, hLb1,
                                hW2, hB2, out);
}
